// MultiHeadGATLayer_94489281070
// MI455X (gfx1250) — matrix operand order repaired and checked by reading, not compiled, not run on hardware
//
#include <hip/hip_runtime.h>
#include <hip/hip_bf16.h>
#include <math.h>

typedef __attribute__((ext_vector_type(16))) _Float16 v16h;
typedef __attribute__((ext_vector_type(8)))  _Float16 v8h;
typedef __attribute__((ext_vector_type(8)))  float    v8f;

#define NDIM    128   // IN_DIM == N_HEADS * D_HEAD
#define NHEADS  4
#define DHEAD   32
#define LN_EPS  1e-5f
#define NEG_SLOPE 0.2f

#define LSTRIDE 136   // padded LDS row stride in f16 elems: 272B = 17*16 (16B aligned,
                      // bank step 4 -> 16 rows at same col hit 16 distinct banks)

// ---------------------------------------------------------------------------
// float atomic max via monotonic int mapping (scores may be negative)
// ---------------------------------------------------------------------------
__device__ __forceinline__ void atomicMaxF(float* addr, float val) {
    if (val >= 0.0f) {
        atomicMax((int*)addr, __float_as_int(val));
    } else {
        atomicMin((unsigned int*)addr, __float_as_uint(val));
    }
}

// ---------------------------------------------------------------------------
// K0: init softmax state + zero the aggregation buffer (d_out)
// ---------------------------------------------------------------------------
__global__ void gat_init_kernel(float* __restrict__ smax, float* __restrict__ denom,
                                float* __restrict__ agg, int nnodes) {
    int i = blockIdx.x * blockDim.x + threadIdx.x;
    int total = nnodes * NDIM;
    if (i < total) agg[i] = 0.0f;
    if (i < nnodes * NHEADS) {
        smax[i]  = -3.402823466e38f;
        denom[i] = 0.0f;
    }
}

// ---------------------------------------------------------------------------
// K1: h = x @ W via v_wmma_f32_16x16x32_f16 (fp32 accumulate).
// Block = 256 threads = 8 waves; each wave owns a 16-row x 128-col stripe.
// LDS staging:
//   Xl : 128x128 x-tile as row-major f16 (stride 136)  -> A frags = 2x ds_load_b128
//   Wt : W transposed (Wt[n][k]) as f16 (stride 136)   -> B frags = 2x ds_load_b128
// ---------------------------------------------------------------------------
__global__ void __launch_bounds__(256)
gat_gemm_kernel(const float* __restrict__ x, const float* __restrict__ W,
                float* __restrict__ h, int nrows) {
    __shared__ _Float16 Wt[NDIM * LSTRIDE];   // ~34 KB
    __shared__ _Float16 Xl[NDIM * LSTRIDE];   // ~34 KB

    const int tid    = threadIdx.x;
    const int mBlock = blockIdx.x * 128;

    // --- stage W^T: coalesced global read, scattered LDS write (one-time) ---
    for (int i = tid; i < NDIM * NDIM; i += 256) {
        int k = i >> 7, n = i & 127;           // W is [K=128][N=128] row-major
        Wt[n * LSTRIDE + k] = (_Float16)W[i];
    }
    // --- stage x tile: coalesced float2 reads, contiguous packed f16 writes ---
    for (int i = tid; i < 128 * 64; i += 256) {     // 64 float2 per row
        int r = i >> 6, c = (i & 63) * 2;
        int gr = mBlock + r;
        if (gr >= nrows) gr = nrows - 1;            // clamp tail rows (stores guarded)
        float2 f = *(const float2*)(x + (size_t)gr * NDIM + c);
        Xl[r * LSTRIDE + c]     = (_Float16)f.x;
        Xl[r * LSTRIDE + c + 1] = (_Float16)f.y;
    }
    __syncthreads();

    const int lane = tid & 31;
    const int wave = tid >> 5;                 // 0..7 -> M sub-tile
    const int half = lane >> 4;
    const int l15  = lane & 15;

    v8f acc[8];
#pragma unroll
    for (int t = 0; t < 8; ++t) acc[t] = (v8f){0,0,0,0,0,0,0,0};

    const _Float16* arow = &Xl[(wave * 16 + l15) * LSTRIDE];

#pragma unroll
    for (int kb = 0; kb < 4; ++kb) {           // K blocks of 32
        // A fragment (16-bit A 16x32, ISA 7.12.2): two contiguous 8-elem runs
        v8h alo = *(const v8h*)(arow + kb * 32 + half * 8);        // K = kb*32+half*8 .. +7
        v8h ahi = *(const v8h*)(arow + kb * 32 + 16 + half * 8);   // K = +16 .. +23
        v16h afrag = __builtin_shufflevector(alo, ahi,
                        0,1,2,3,4,5,6,7, 8,9,10,11,12,13,14,15);
#pragma unroll
        for (int nt = 0; nt < 8; ++nt) {       // 8 N-tiles of 16 cols
            // B fragment: lane holds col n = nt*16+l15; K = kb*32 + half*8 + 0..7, then +16 .. +23
            const _Float16* bp = &Wt[(nt * 16 + l15) * LSTRIDE + kb * 32 + half * 8];
            v8h blo = *(const v8h*)(bp);
            v8h bhi = *(const v8h*)(bp + 16);
            v16h bfrag = __builtin_shufflevector(blo, bhi,
                            0,1,2,3,4,5,6,7, 8,9,10,11,12,13,14,15);
            acc[nt] = __builtin_amdgcn_wmma_f32_16x16x32_f16(
                false, afrag, false, bfrag, (short)0, acc[nt], false, false);
        }
    }

    // --- store: C/D layout: VGPR v -> row M = v + half*8, col N = l15 ---
#pragma unroll
    for (int nt = 0; nt < 8; ++nt) {
        int ncol = nt * 16 + l15;
#pragma unroll
        for (int v = 0; v < 8; ++v) {
            int m = mBlock + wave * 16 + half * 8 + v;
            if (m < nrows) h[(size_t)m * NDIM + ncol] = acc[nt][v];
        }
    }
}

// ---------------------------------------------------------------------------
// K2: per (edge, head): s = <h_src, a_src> + <h_dst, a_dst>, LeakyReLU,
//     atomic segment-max into smax[dst]. h is L2-resident (51 MB < 192 MB).
// ---------------------------------------------------------------------------
__global__ void gat_score_kernel(const float* __restrict__ h,
                                 const int* __restrict__ src, const int* __restrict__ dst,
                                 const float* __restrict__ a_src, const float* __restrict__ a_dst,
                                 float* __restrict__ s, float* __restrict__ smax, int nedges) {
    int idx = blockIdx.x * blockDim.x + threadIdx.x;
    if (idx >= nedges * NHEADS) return;
    int e  = idx >> 2;
    int hh = idx & 3;
    int si = src[e], di = dst[e];
    const float4* hs = (const float4*)(h + (size_t)si * NDIM + hh * DHEAD);
    const float4* hd = (const float4*)(h + (size_t)di * NDIM + hh * DHEAD);
    const float4* as = (const float4*)a_src;
    const float4* ad = (const float4*)a_dst;
    float acc = 0.0f;
#pragma unroll
    for (int q = 0; q < DHEAD / 4; ++q) {
        float4 a = hs[q], b = hd[q], wa = as[q], wb = ad[q];
        acc += a.x * wa.x + a.y * wa.y + a.z * wa.z + a.w * wa.w;
        acc += b.x * wb.x + b.y * wb.y + b.z * wb.z + b.w * wb.w;
    }
    float sv = acc > 0.0f ? acc : NEG_SLOPE * acc;
    s[idx] = sv;
    atomicMaxF(&smax[di * NHEADS + hh], sv);
}

// ---------------------------------------------------------------------------
// K3: e = exp(s - smax[dst]); denom[dst] += e  (segment-sum via L2 atomics)
// ---------------------------------------------------------------------------
__global__ void gat_exp_kernel(const int* __restrict__ dst,
                               float* __restrict__ s, const float* __restrict__ smax,
                               float* __restrict__ denom, int nedges) {
    int idx = blockIdx.x * blockDim.x + threadIdx.x;
    if (idx >= nedges * NHEADS) return;
    int e  = idx >> 2;
    int hh = idx & 3;
    int di = dst[e];
    float ev = __expf(s[idx] - smax[di * NHEADS + hh]);
    s[idx] = ev;
    atomicAdd(&denom[di * NHEADS + hh], ev);
}

// ---------------------------------------------------------------------------
// K4: alpha = e / denom[dst]; agg[dst] += alpha * h_src  (scatter-add, L2 atomics)
// ---------------------------------------------------------------------------
__global__ void gat_agg_kernel(const float* __restrict__ h,
                               const int* __restrict__ src, const int* __restrict__ dst,
                               const float* __restrict__ s, const float* __restrict__ denom,
                               float* __restrict__ agg, int nedges) {
    int idx = blockIdx.x * blockDim.x + threadIdx.x;
    if (idx >= nedges * NHEADS) return;
    int e  = idx >> 2;
    int hh = idx & 3;
    int si = src[e], di = dst[e];
    float alpha = s[idx] / denom[di * NHEADS + hh];
    const float* hs = h + (size_t)si * NDIM + hh * DHEAD;
    float* op = agg + (size_t)di * NDIM + hh * DHEAD;
#pragma unroll
    for (int d = 0; d < DHEAD; ++d)
        atomicAdd(&op[d], alpha * hs[d]);
}

// ---------------------------------------------------------------------------
// K5: out = ELU(LayerNorm(agg + x)); one wave32 per node, 4 dims per lane.
// ---------------------------------------------------------------------------
__global__ void __launch_bounds__(256)
gat_finalize_kernel(const float* __restrict__ x,
                    const float* __restrict__ gamma, const float* __restrict__ beta,
                    float* __restrict__ out, int nnodes) {
    int node = blockIdx.x * (blockDim.x >> 5) + (threadIdx.x >> 5);
    if (node >= nnodes) return;
    int lane = threadIdx.x & 31;

    float4 v  = ((const float4*)(out + (size_t)node * NDIM))[lane];
    float4 xv = ((const float4*)(x   + (size_t)node * NDIM))[lane];
    v.x += xv.x; v.y += xv.y; v.z += xv.z; v.w += xv.w;

    float sum = v.x + v.y + v.z + v.w;
#pragma unroll
    for (int o = 16; o > 0; o >>= 1) sum += __shfl_xor(sum, o, 32);
    float mu = sum * (1.0f / NDIM);

    float dx = v.x - mu, dy = v.y - mu, dz = v.z - mu, dw = v.w - mu;
    float vs = dx * dx + dy * dy + dz * dz + dw * dw;
#pragma unroll
    for (int o = 16; o > 0; o >>= 1) vs += __shfl_xor(vs, o, 32);
    float inv = rsqrtf(vs * (1.0f / NDIM) + LN_EPS);

    float4 g = ((const float4*)gamma)[lane];
    float4 b = ((const float4*)beta)[lane];
    float4 y;
    y.x = dx * inv * g.x + b.x;
    y.y = dy * inv * g.y + b.y;
    y.z = dz * inv * g.z + b.z;
    y.w = dw * inv * g.w + b.w;
    y.x = y.x > 0.0f ? y.x : expm1f(y.x);
    y.y = y.y > 0.0f ? y.y : expm1f(y.y);
    y.z = y.z > 0.0f ? y.z : expm1f(y.z);
    y.w = y.w > 0.0f ? y.w : expm1f(y.w);
    ((float4*)(out + (size_t)node * NDIM))[lane] = y;
}

// ---------------------------------------------------------------------------
// launch
// ---------------------------------------------------------------------------
extern "C" void kernel_launch(void* const* d_in, const int* in_sizes, int n_in,
                              void* d_out, int out_size, void* d_ws, size_t ws_size,
                              hipStream_t stream) {
    const float* x      = (const float*)d_in[0];
    const int*   eidx   = (const int*)d_in[1];
    const float* W      = (const float*)d_in[2];
    const float* a_src  = (const float*)d_in[3];
    const float* a_dst  = (const float*)d_in[4];
    const float* gamma  = (const float*)d_in[5];
    const float* beta   = (const float*)d_in[6];
    float*       out    = (float*)d_out;

    const int nnodes = in_sizes[0] / NDIM;     // 100000
    const int nedges = in_sizes[1] / 2;        // 800000
    const int* src = eidx;
    const int* dst = eidx + nedges;

    // workspace layout (floats): h [N*128] | s [E*4] | smax [N*4] | denom [N*4]
    float* h     = (float*)d_ws;
    float* s     = h + (size_t)nnodes * NDIM;
    float* smax  = s + (size_t)nedges * NHEADS;
    float* denom = smax + (size_t)nnodes * NHEADS;

    const int B = 256;

    // K0: init smax/denom + zero agg (d_out)
    {
        int total = nnodes * NDIM;
        gat_init_kernel<<<(total + B - 1) / B, B, 0, stream>>>(smax, denom, out, nnodes);
    }
    // K1: WMMA GEMM h = x @ W
    {
        int nblocks = (nnodes + 127) / 128;
        gat_gemm_kernel<<<nblocks, B, 0, stream>>>(x, W, h, nnodes);
    }
    // K2..K4: edge phase (E*4 threads each)
    {
        int total = nedges * NHEADS;
        int g = (total + B - 1) / B;
        gat_score_kernel<<<g, B, 0, stream>>>(h, src, dst, a_src, a_dst, s, smax, nedges);
        gat_exp_kernel<<<g, B, 0, stream>>>(dst, s, smax, denom, nedges);
        gat_agg_kernel<<<g, B, 0, stream>>>(h, src, dst, s, denom, out, nedges);
    }
    // K5: residual + LayerNorm + ELU (wave32 per node)
    {
        int nodesPerBlock = B / 32;
        gat_finalize_kernel<<<(nnodes + nodesPerBlock - 1) / nodesPerBlock, B, 0, stream>>>(
            x, gamma, beta, out, nnodes);
    }
}
